// Attention_27144193311525
// MI455X (gfx1250) — matrix operand order repaired and checked by reading, not compiled, not run on hardware
//
#include <hip/hip_runtime.h>
#include <hip/hip_bf16.h>
#include <math.h>

typedef _Float16 v16h __attribute__((ext_vector_type(16)));
typedef _Float16 v8h  __attribute__((ext_vector_type(8)));
typedef float    v8f  __attribute__((ext_vector_type(8)));

#define S_LEN 2048
#define D_DIM 64
#define H_NUM 8
#define BH_NUM 16
#define MTILE 64     // query rows per workgroup
#define KTILE 32     // key rows per iteration (2 x 16-wide WMMA tiles)

// exp2-domain constants:  s = exp(c)/8  ->  u = s*log2(e) = 2^(c*log2e + KU)
#define L2E   1.4426950408889634f
#define KU    (-2.4712336270551023f)   // log2(log2(e)) - 3
#define UMASK (-1.4426950e9f)          // (-1e9) * log2(e)

static __device__ __forceinline__ float exp2f_fast(float x) {
    return __builtin_amdgcn_exp2f(x);      // raw v_exp_f32
}

// merge two (max, sum) softmax partials with a single exp
static __device__ __forceinline__ void merge_ml(float& M, float& L,
                                                float om, float ol) {
    const float nM = fmaxf(M, om);
    const float e  = exp2f_fast(fminf(M, om) - nM);
    L = (M >= om) ? fmaf(ol, e, L) : fmaf(L, e, ol);
    M = nM;
}

// ---------------- prologue: K fp32 -> f16 (same layout) ----------------
__global__ __launch_bounds__(256)
void cvt_k_kernel(const float* __restrict__ K, _Float16* __restrict__ Kh)
{
    const size_t i = ((size_t)blockIdx.x * 256 + threadIdx.x) * 8;
    const float4 a = *(const float4*)(K + i);
    const float4 b = *(const float4*)(K + i + 4);
    v8h o;
    o[0] = (_Float16)a.x; o[1] = (_Float16)a.y;
    o[2] = (_Float16)a.z; o[3] = (_Float16)a.w;
    o[4] = (_Float16)b.x; o[5] = (_Float16)b.y;
    o[6] = (_Float16)b.z; o[7] = (_Float16)b.w;
    *(v8h*)(Kh + i) = o;
}

// ------------- prologue: V fp32 [bh][s][d] -> f16 [bh][d][s] -----------
__global__ __launch_bounds__(256)
void cvt_vt_kernel(const float* __restrict__ V, _Float16* __restrict__ Vt)
{
    const int t  = blockIdx.x * 256 + threadIdx.x;
    const int s  = t & (S_LEN - 1);
    const int d  = (t >> 11) & (D_DIM - 1);
    const int bh = t >> 17;
    Vt[t] = (_Float16)V[((size_t)bh * S_LEN + s) * D_DIM + d];
}

// ------------------------------ main -----------------------------------
__global__ __launch_bounds__(256)
void attn_exp_softmax_kernel(const float* __restrict__ Q,
                             const _Float16* __restrict__ Kh,
                             const _Float16* __restrict__ Vt,
                             const int*   __restrict__ mask,
                             float* __restrict__ out,
                             float* __restrict__ pattn)
{
    const int bh = blockIdx.y;              // 0..15  (b*H + h)
    const int b  = bh / H_NUM;
    const int q0 = blockIdx.x * MTILE;

    const int tid  = threadIdx.x;
    const int lane = tid & 31;
    const int w    = tid >> 5;              // wave id 0..7
    const int mw   = w >> 1;                // M-block 0..3
    const int nw   = w & 1;                 // N-half  0..1
    const int half = lane >> 4;             // WMMA lane-half
    const int l15  = lane & 15;

    const size_t qkv_base = (size_t)bh * S_LEN * D_DIM;
    const float*    Qb = Q  + qkv_base;
    const _Float16* Kb = Kh + qkv_base;
    const _Float16* Vb = Vt + (size_t)bh * D_DIM * S_LEN;   // [d][s]
    const int*      Mb = mask + (size_t)b * S_LEN;

    __shared__ __align__(16) _Float16 Ps[MTILE][40];  // 80B row stride: 16B aligned,
                                                      // 20-bank step -> conflict-free
    __shared__ float sM[2][MTILE];
    __shared__ float sL[2][MTILE];
    __shared__ float rowM[MTILE];
    __shared__ float rowInvL[MTILE];

    // ---- per-lane mask bitmask: this lane only ever tests col = 32*t + nw*16 + l15
    unsigned long long mbits = 0ull;
    for (int t = 0; t < S_LEN / KTILE; ++t)
        mbits |= (unsigned long long)(Mb[t * KTILE + nw * 16 + l15] != 0) << t;

    // ---- Q A-fragments (fp32 load + cvt, once per workgroup) ----
    v16h qa0, qa1;
    {
        const int mrow = q0 + mw * 16 + l15;
        const float* qrow = Qb + (size_t)mrow * D_DIM;
#pragma unroll
        for (int j = 0; j < 16; ++j) {
            const int k = j + 16 * half;
            qa0[j] = (_Float16)qrow[k];
            qa1[j] = (_Float16)qrow[32 + k];
        }
    }

    // one 16x16 f32 QK^T tile at key-tile k0 (D=64 -> 2 WMMAs, f16 b128 loads)
    auto qk_tile = [&](int k0) -> v8f {
        const int ncol = k0 + nw * 16 + l15;
        const _Float16* krow = Kb + (size_t)ncol * D_DIM;
        const v16h kb0 = *(const v16h*)(krow + 16 * half);
        const v16h kb1 = *(const v16h*)(krow + 32 + 16 * half);
        if (k0 + KTILE < S_LEN)
            __builtin_prefetch(Kb + (size_t)(ncol + KTILE) * D_DIM, 0, 1);
        v8f c = {};
        c = __builtin_amdgcn_wmma_f32_16x16x32_f16(false, qa0, false, kb0,
                                                   (short)0, c, false, false);
        c = __builtin_amdgcn_wmma_f32_16x16x32_f16(false, qa1, false, kb1,
                                                   (short)0, c, false, false);
        return c;
    };

    // ===== Pass 1: per-LANE online (max, sum); 2 trans/elem, no cross-lane ops =====
    float M_[8], L_[8];
#pragma unroll
    for (int r = 0; r < 8; ++r) { M_[r] = -__builtin_huge_valf(); L_[r] = 0.0f; }

    for (int k0 = 0; k0 < S_LEN; k0 += KTILE) {
        const v8f c = qk_tile(k0);
        const bool mk = (mbits >> (k0 >> 5)) & 1;
#pragma unroll
        for (int r = 0; r < 8; ++r) {
            float u = exp2f_fast(fmaf(c[r], L2E, KU));
            u = mk ? u : UMASK;
            // single-exp online update: one of the two rescale exponents is 0
            const float hi = fmaxf(M_[r], u);
            const float e  = exp2f_fast(fminf(M_[r], u) - hi);
            L_[r] = (u > M_[r]) ? fmaf(L_[r], e, 1.0f) : (L_[r] + e);
            M_[r] = hi;
        }
    }

    // butterfly merge of (M,L) across the 16 lanes holding each row (once)
#pragma unroll
    for (int r = 0; r < 8; ++r) {
#pragma unroll
        for (int st = 1; st <= 8; st <<= 1) {
            const float om = __shfl_xor(M_[r], st, 32);
            const float ol = __shfl_xor(L_[r], st, 32);
            merge_ml(M_[r], L_[r], om, ol);
        }
    }

    // merge the two N-wave partials per row through LDS
    if (l15 == 0) {
#pragma unroll
        for (int r = 0; r < 8; ++r) {
            const int row = mw * 16 + half * 8 + r;
            sM[nw][row] = M_[r];
            sL[nw][row] = L_[r];
        }
    }
    __syncthreads();
    if (tid < MTILE) {
        float m = sM[0][tid], l = sL[0][tid];
        merge_ml(m, l, sM[1][tid], sL[1][tid]);
        rowM[tid]    = m;
        rowInvL[tid] = 1.0f / l;
    }
    __syncthreads();

    float rM[8], rIL[8];
#pragma unroll
    for (int r = 0; r < 8; ++r) {
        const int row = mw * 16 + half * 8 + r;
        rM[r]  = rowM[row];
        rIL[r] = rowInvL[row];
    }
    __syncthreads();

    // =================== Pass 2: p_attn writeback + P.V ===================
    v8f acc0 = {}, acc1 = {};
    float* pb = pattn + (size_t)bh * S_LEN * S_LEN;

    for (int k0 = 0; k0 < S_LEN; k0 += KTILE) {
        const v8f c = qk_tile(k0);
        const int ncol = k0 + nw * 16 + l15;
        const bool mk = (mbits >> (k0 >> 5)) & 1;
#pragma unroll
        for (int r = 0; r < 8; ++r) {
            float u = exp2f_fast(fmaf(c[r], L2E, KU));
            u = mk ? u : UMASK;
            const float p = exp2f_fast(u - rM[r]) * rIL[r];
            const int row = q0 + mw * 16 + half * 8 + r;
            __builtin_nontemporal_store(p, pb + (size_t)row * S_LEN + ncol);
            Ps[mw * 16 + half * 8 + r][nw * 16 + l15] = (_Float16)p;
        }
        __syncthreads();

        // A-fragment of P: two contiguous 16B LDS vector loads per lane
        v16h pa;
        {
            const _Float16* prow = &Ps[mw * 16 + l15][0];
            const v8h plo = *(const v8h*)(prow + 16 * half);
            const v8h phi = *(const v8h*)(prow + 16 * half + 8);
#pragma unroll
            for (int j = 0; j < 8; ++j) { pa[j] = plo[j]; pa[j + 8] = phi[j]; }
        }
        // B-fragments of V from transposed f16 V: contiguous 32B vector loads
        const _Float16* vr0 = Vb + (size_t)(nw * 32 + l15) * S_LEN + k0 + 16 * half;
        const v16h vb0 = *(const v16h*)vr0;
        const v16h vb1 = *(const v16h*)(vr0 + (size_t)16 * S_LEN);

        acc0 = __builtin_amdgcn_wmma_f32_16x16x32_f16(false, pa, false, vb0,
                                                      (short)0, acc0, false, false);
        acc1 = __builtin_amdgcn_wmma_f32_16x16x32_f16(false, pa, false, vb1,
                                                      (short)0, acc1, false, false);
        __syncthreads();
    }

    // ---- write out (B,H,S,D) ----
    float* ob = out + qkv_base;
#pragma unroll
    for (int r = 0; r < 8; ++r) {
        const int row = q0 + mw * 16 + half * 8 + r;
        ob[(size_t)row * D_DIM + nw * 32 + l15]      = acc0[r];
        ob[(size_t)row * D_DIM + nw * 32 + 16 + l15] = acc1[r];
    }
}

extern "C" void kernel_launch(void* const* d_in, const int* in_sizes, int n_in,
                              void* d_out, int out_size, void* d_ws, size_t ws_size,
                              hipStream_t stream) {
    const float* Q    = (const float*)d_in[0];
    const float* K    = (const float*)d_in[1];
    const float* V    = (const float*)d_in[2];
    const int*   mask = (const int*)d_in[3];

    float* out   = (float*)d_out;
    float* pattn = out + (size_t)BH_NUM * S_LEN * D_DIM;

    _Float16* Kh = (_Float16*)d_ws;                          // 4 MB
    _Float16* Vt = Kh + (size_t)BH_NUM * S_LEN * D_DIM;      // 4 MB

    const int total = BH_NUM * S_LEN * D_DIM;                // 2M elements
    cvt_k_kernel <<<total / 8 / 256, 256, 0, stream>>>(K, Kh);
    cvt_vt_kernel<<<total / 256,     256, 0, stream>>>(V, Vt);

    dim3 grid(S_LEN / MTILE, BH_NUM);                        // 32 x 16
    attn_exp_softmax_kernel<<<grid, 256, 0, stream>>>(Q, Kh, Vt, mask, out, pattn);
}
